// MoEBlock_6992206758281
// MI455X (gfx1250) — matrix operand order repaired and checked by reading, not compiled, not run on hardware
//
#include <hip/hip_runtime.h>
#include <hip/hip_bf16.h>
#include <math.h>

// ---------------- problem constants ----------------
#define D_DIM   1024
#define FF_DIM  4096
#define NE      8
#define NTOK    4096          // B*L
#define CAP     4096          // per-expert row capacity (worst case)
#define TOTROWS 12288         // 2*NTOK expert rows + NTOK shared rows

// ---------------- GEMM tiling ----------------
#define BM      128
#define BN      128
#define BK      32
#define APITCH  40            // bf16 elements; multiple of 8 -> 16B-aligned frags
#define BPITCH  40

typedef __attribute__((ext_vector_type(16))) __bf16 v16bf;
typedef __attribute__((ext_vector_type(8)))  __bf16 bf16x8;
typedef __attribute__((ext_vector_type(8)))  float  v8f;
typedef __attribute__((ext_vector_type(4)))  int    v4i;

union FragAB { v16bf v; bf16x8 h[2]; };

// ---------------- CDNA5 async global->LDS path ----------------
#if defined(__gfx1250__) && __has_builtin(__builtin_amdgcn_global_load_async_to_lds_b128)
#define HAVE_ASYNC 1
#else
#define HAVE_ASYNC 0
#endif

__device__ __forceinline__ void async_copy_b128(const __bf16* g, __bf16* l) {
#if HAVE_ASYNC
    __builtin_amdgcn_global_load_async_to_lds_b128(
        (__attribute__((address_space(1))) v4i*)g,
        (__attribute__((address_space(3))) v4i*)l, 0, 0);
#else
    *(bf16x8*)l = *(const bf16x8*)g;
#endif
}

__device__ __forceinline__ void async_wait() {
#if HAVE_ASYNC
#if __has_builtin(__builtin_amdgcn_s_wait_asynccnt)
    __builtin_amdgcn_s_wait_asynccnt(0);
#else
    asm volatile("s_wait_asynccnt 0" ::: "memory");
#endif
#endif
}

__device__ __forceinline__ v8f zero_v8f() {
    v8f z = {0.f, 0.f, 0.f, 0.f, 0.f, 0.f, 0.f, 0.f};
    return z;
}

__device__ __forceinline__ float gelu_exact(float t) {
    return 0.5f * t * (1.0f + erff(t * 0.70710678118654752f));
}

__device__ __forceinline__ v8f wmma_bf16(const FragAB& a, const FragAB& b, v8f c) {
    return __builtin_amdgcn_wmma_f32_16x16x32_bf16(
        false, a.v, false, b.v, (short)0, c, false, false);
}

// ============================================================
// Kernel 0: zero the per-expert atomic counters
// ============================================================
__global__ void zero_kernel(int* __restrict__ cnt) {
    if (threadIdx.x < NE) cnt[threadIdx.x] = 0;
}

// ============================================================
// Kernel 1: x (fp32) -> xb (bf16), once
// ============================================================
__global__ __launch_bounds__(256) void xcvt_kernel(
    const float* __restrict__ x, __bf16* __restrict__ xb)
{
    size_t i = ((size_t)blockIdx.x * 256 + threadIdx.x) * 8;
    const float4* s = (const float4*)(x + i);
    float4 f0 = s[0], f1 = s[1];
    bf16x8 v;
    v[0]=(__bf16)f0.x; v[1]=(__bf16)f0.y; v[2]=(__bf16)f0.z; v[3]=(__bf16)f0.w;
    v[4]=(__bf16)f1.x; v[5]=(__bf16)f1.y; v[6]=(__bf16)f1.z; v[7]=(__bf16)f1.w;
    *(bf16x8*)(xb + i) = v;
}

// ============================================================
// Kernel 2: gating — one wave32 per token
// ============================================================
__global__ __launch_bounds__(256) void gate_kernel(
    const float* __restrict__ x, const float* __restrict__ Wg,
    int* __restrict__ cnt, int* __restrict__ tokList, float* __restrict__ vList,
    int* __restrict__ tokE, int* __restrict__ tokR)
{
    const int lane = threadIdx.x & 31;
    const int wid  = threadIdx.x >> 5;
    const int t    = blockIdx.x * 8 + wid;

    float acc[NE];
#pragma unroll
    for (int e = 0; e < NE; ++e) acc[e] = 0.f;

    const float* xrow = x + (size_t)t * D_DIM;
    for (int d = lane; d < D_DIM; d += 32) {
        float xv = xrow[d];
        const float4* wg = (const float4*)(Wg + (size_t)d * NE);
        float4 a = wg[0], b = wg[1];
        acc[0] += xv * a.x; acc[1] += xv * a.y;
        acc[2] += xv * a.z; acc[3] += xv * a.w;
        acc[4] += xv * b.x; acc[5] += xv * b.y;
        acc[6] += xv * b.z; acc[7] += xv * b.w;
    }
#pragma unroll
    for (int e = 0; e < NE; ++e) {
#pragma unroll
        for (int off = 16; off > 0; off >>= 1)
            acc[e] += __shfl_xor(acc[e], off, 32);
    }

    if (lane == 0) {
        float m = acc[0];
#pragma unroll
        for (int e = 1; e < NE; ++e) m = fmaxf(m, acc[e]);
        float p[NE];
#pragma unroll
        for (int e = 0; e < NE; ++e) p[e] = __expf(acc[e] - m);

        int i1 = 0; float v1 = p[0];
#pragma unroll
        for (int e = 1; e < NE; ++e) if (p[e] > v1) { v1 = p[e]; i1 = e; }
        int i2 = (i1 == 0) ? 1 : 0; float v2 = p[i2];
#pragma unroll
        for (int e = 0; e < NE; ++e)
            if (e != i1 && p[e] > v2) { v2 = p[e]; i2 = e; }

        float s  = v1 + v2;       // softmax denominator cancels in the ratio
        float w1 = v1 / s;
        float w2 = v2 / s;

        int r1 = atomicAdd(&cnt[i1], 1);
        tokList[i1 * CAP + r1] = t;
        vList  [i1 * CAP + r1] = w1;
        int r2 = atomicAdd(&cnt[i2], 1);
        tokList[i2 * CAP + r2] = t;
        vList  [i2 * CAP + r2] = w2;

        tokE[2 * t]     = i1; tokR[2 * t]     = r1;
        tokE[2 * t + 1] = i2; tokR[2 * t + 1] = r2;
    }
}

// ============================================================
// Kernel 3: exclusive scan of expert counts
// ============================================================
__global__ void scan_kernel(const int* __restrict__ cnt, int* __restrict__ base) {
    if (threadIdx.x == 0) {
        int b = 0;
#pragma unroll
        for (int e = 0; e < NE; ++e) { base[e] = b; b += cnt[e]; }
        base[NE] = b;
    }
}

// ============================================================
// Kernel 4: h = gelu(x_rows @ W1[e] + b1[e])  (bf16 out)
// BM=128,BN=128,BK=32, double-buffered LDS, async A staging
// ============================================================
__global__ __launch_bounds__(256) void ffn1_kernel(
    const __bf16* __restrict__ xb,
    const float* __restrict__ W1,  const float* __restrict__ b1,
    const float* __restrict__ Ws1, const float* __restrict__ bs1,
    const int* __restrict__ cnt, const int* __restrict__ base,
    const int* __restrict__ tokList,
    __bf16* __restrict__ h)
{
    const int e  = blockIdx.z;
    const int rt = blockIdx.y;
    const int nn = blockIdx.x * BN;

    int n, rowBase;
    const float* Wp;
    const float* bp;
    if (e < NE) {
        n = cnt[e];
        if (rt * BM >= n) return;                 // uniform early exit
        rowBase = base[e];
        Wp = W1 + (size_t)e * D_DIM * FF_DIM;
        bp = b1 + (size_t)e * FF_DIM;
    } else {
        n = NTOK; rowBase = base[NE]; Wp = Ws1; bp = bs1;
    }

    __shared__ __bf16 As[2][BM][APITCH];
    __shared__ __bf16 Bs[2][BN][BPITCH];
    __shared__ int tokLds[BM];

    const int tid = threadIdx.x;
    if (tid < BM) {
        int r  = rt * BM + tid;
        int rc = (r < n) ? r : (n - 1);
        tokLds[tid] = (e < NE) ? tokList[e * CAP + rc] : rc;
    }
    __syncthreads();

    // staging roles
    const int arow = tid >> 1, aseg = tid & 1;     // A: 2 threads / row (2x16B)
    const int bcol = tid & 127, bkg = tid >> 7;    // B: col + K-halftile
    const __bf16* aSrc = xb + (size_t)tokLds[arow] * D_DIM + aseg * 16;
    const float*  bSrc = Wp + (size_t)(bkg * 16) * FF_DIM + nn + bcol;

    const int lane = tid & 31, w = tid >> 5;
    const int l16  = lane & 15, half = lane >> 4;
    const int mt0  = (w & 3) * 2;                  // two A row-tiles per wave
    const int g    = w >> 2;                       // column group (0/1)

    v8f acc[8];
#pragma unroll
    for (int i = 0; i < 8; ++i) acc[i] = zero_v8f();

    auto stageA = [&](int buf, int kk) {
        const __bf16* s = aSrc + kk;
        __bf16* d = &As[buf][arow][aseg * 16];
        async_copy_b128(s, d);
        async_copy_b128(s + 8, d + 8);
    };
    auto stageB = [&](int buf, int kk) {
        const float* s = bSrc + (size_t)kk * FF_DIM;
        union { __bf16 s16[16]; bf16x8 v2[2]; } tmp;
#pragma unroll
        for (int j = 0; j < 16; ++j) tmp.s16[j] = (__bf16)s[(size_t)j * FF_DIM];
        *(bf16x8*)&Bs[buf][bcol][bkg * 16]     = tmp.v2[0];
        *(bf16x8*)&Bs[buf][bcol][bkg * 16 + 8] = tmp.v2[1];
        if (kk + 2 * BK < D_DIM)                   // stream hint 2 tiles ahead
            __builtin_prefetch(s + (size_t)(2 * BK) * FF_DIM, 0, 0);
    };

    stageA(0, 0); stageB(0, 0);
    async_wait();
    __syncthreads();

    const int nsteps = D_DIM / BK;
    for (int st = 0; st < nsteps; ++st) {
        const int cur = st & 1;
        if (st + 1 < nsteps) { stageA(cur ^ 1, (st + 1) * BK); stageB(cur ^ 1, (st + 1) * BK); }

        // ---- all fragments into distinct regs, then 8 back-to-back WMMAs ----
        FragAB a0, a1, bfr[4];
        a0.h[0] = *(const bf16x8*)&As[cur][mt0 * 16 + l16][half * 8];
        a0.h[1] = *(const bf16x8*)&As[cur][mt0 * 16 + l16][half * 8 + 16];
        a1.h[0] = *(const bf16x8*)&As[cur][(mt0 + 1) * 16 + l16][half * 8];
        a1.h[1] = *(const bf16x8*)&As[cur][(mt0 + 1) * 16 + l16][half * 8 + 16];
#pragma unroll
        for (int j = 0; j < 4; ++j) {
            int ncol = (g * 4 + j) * 16 + l16;
            bfr[j].h[0] = *(const bf16x8*)&Bs[cur][ncol][half * 8];
            bfr[j].h[1] = *(const bf16x8*)&Bs[cur][ncol][half * 8 + 16];
        }
#pragma unroll
        for (int j = 0; j < 4; ++j) {
            acc[j]     = wmma_bf16(a0, bfr[j], acc[j]);
            acc[4 + j] = wmma_bf16(a1, bfr[j], acc[4 + j]);
        }

        async_wait();
        __syncthreads();
    }

    // ---- epilogue: bias + exact GELU, store bf16 ----
#pragma unroll
    for (int i = 0; i < 8; ++i) {
        int mtile = mt0 + (i >> 2);
        int j     = i & 3;
        int col   = nn + (g * 4 + j) * 16 + l16;
        float bias = bp[col];
#pragma unroll
        for (int r = 0; r < 8; ++r) {
            int m  = mtile * 16 + half * 8 + r;
            int rr = rt * BM + m;
            if (rr < n) {
                float tv = acc[i][r] + bias;
                h[(size_t)(rowBase + rr) * FF_DIM + col] = (__bf16)gelu_exact(tv);
            }
        }
    }
}

// ============================================================
// Kernel 5: y = combine_v * (h_rows @ W2[e] + b2[e])  (fp32 out)
// ============================================================
__global__ __launch_bounds__(256) void ffn2_kernel(
    const __bf16* __restrict__ h,
    const float* __restrict__ W2,  const float* __restrict__ b2,
    const float* __restrict__ Ws2, const float* __restrict__ bs2,
    const int* __restrict__ cnt, const int* __restrict__ base,
    const float* __restrict__ vList,
    float* __restrict__ y)
{
    const int e  = blockIdx.z;
    const int rt = blockIdx.y;
    const int nn = blockIdx.x * BN;

    int n, rowBase;
    const float* Wp;
    const float* bp;
    if (e < NE) {
        n = cnt[e];
        if (rt * BM >= n) return;
        rowBase = base[e];
        Wp = W2 + (size_t)e * FF_DIM * D_DIM;
        bp = b2 + (size_t)e * D_DIM;
    } else {
        n = NTOK; rowBase = base[NE]; Wp = Ws2; bp = bs2;
    }

    __shared__ __bf16 As[2][BM][APITCH];
    __shared__ __bf16 Bs[2][BN][BPITCH];
    __shared__ float vLds[BM];

    const int tid = threadIdx.x;
    if (tid < BM) {
        int r = rt * BM + tid;
        vLds[tid] = (e < NE) ? ((r < n) ? vList[e * CAP + r] : 0.f) : 1.0f;
    }
    __syncthreads();

    const int arow = tid >> 1, aseg = tid & 1;
    const int bcol = tid & 127, bkg = tid >> 7;
    int arc = rt * BM + arow; if (arc >= n) arc = n - 1;
    const __bf16* aSrc = h + (size_t)(rowBase + arc) * FF_DIM + aseg * 16;
    const float*  bSrc = Wp + (size_t)(bkg * 16) * D_DIM + nn + bcol;

    const int lane = tid & 31, w = tid >> 5;
    const int l16  = lane & 15, half = lane >> 4;
    const int mt0  = (w & 3) * 2;
    const int g    = w >> 2;

    v8f acc[8];
#pragma unroll
    for (int i = 0; i < 8; ++i) acc[i] = zero_v8f();

    auto stageA = [&](int buf, int kk) {
        const __bf16* s = aSrc + kk;
        __bf16* d = &As[buf][arow][aseg * 16];
        async_copy_b128(s, d);
        async_copy_b128(s + 8, d + 8);
    };
    auto stageB = [&](int buf, int kk) {
        const float* s = bSrc + (size_t)kk * D_DIM;
        union { __bf16 s16[16]; bf16x8 v2[2]; } tmp;
#pragma unroll
        for (int j = 0; j < 16; ++j) tmp.s16[j] = (__bf16)s[(size_t)j * D_DIM];
        *(bf16x8*)&Bs[buf][bcol][bkg * 16]     = tmp.v2[0];
        *(bf16x8*)&Bs[buf][bcol][bkg * 16 + 8] = tmp.v2[1];
        if (kk + 2 * BK < FF_DIM)
            __builtin_prefetch(s + (size_t)(2 * BK) * D_DIM, 0, 0);
    };

    stageA(0, 0); stageB(0, 0);
    async_wait();
    __syncthreads();

    const int nsteps = FF_DIM / BK;
    for (int st = 0; st < nsteps; ++st) {
        const int cur = st & 1;
        if (st + 1 < nsteps) { stageA(cur ^ 1, (st + 1) * BK); stageB(cur ^ 1, (st + 1) * BK); }

        FragAB a0, a1, bfr[4];
        a0.h[0] = *(const bf16x8*)&As[cur][mt0 * 16 + l16][half * 8];
        a0.h[1] = *(const bf16x8*)&As[cur][mt0 * 16 + l16][half * 8 + 16];
        a1.h[0] = *(const bf16x8*)&As[cur][(mt0 + 1) * 16 + l16][half * 8];
        a1.h[1] = *(const bf16x8*)&As[cur][(mt0 + 1) * 16 + l16][half * 8 + 16];
#pragma unroll
        for (int j = 0; j < 4; ++j) {
            int ncol = (g * 4 + j) * 16 + l16;
            bfr[j].h[0] = *(const bf16x8*)&Bs[cur][ncol][half * 8];
            bfr[j].h[1] = *(const bf16x8*)&Bs[cur][ncol][half * 8 + 16];
        }
#pragma unroll
        for (int j = 0; j < 4; ++j) {
            acc[j]     = wmma_bf16(a0, bfr[j], acc[j]);
            acc[4 + j] = wmma_bf16(a1, bfr[j], acc[4 + j]);
        }

        async_wait();
        __syncthreads();
    }

#pragma unroll
    for (int i = 0; i < 8; ++i) {
        int mtile = mt0 + (i >> 2);
        int j     = i & 3;
        int col   = nn + (g * 4 + j) * 16 + l16;
        float bias = bp[col];
#pragma unroll
        for (int r = 0; r < 8; ++r) {
            int m  = mtile * 16 + half * 8 + r;
            int rr = rt * BM + m;
            if (rr < n) {
                y[(size_t)(rowBase + rr) * D_DIM + col] =
                    vLds[m] * (acc[i][r] + bias);
            }
        }
    }
}

// ============================================================
// Kernel 6: out[t] = y[rowA] + y[rowB] + y[sharedRow]
// ============================================================
__global__ __launch_bounds__(256) void combine_kernel(
    const float* __restrict__ y, const int* __restrict__ base,
    const int* __restrict__ tokE, const int* __restrict__ tokR,
    float* __restrict__ out)
{
    const int t  = blockIdx.x;
    const int e1 = tokE[2 * t],     e2 = tokE[2 * t + 1];
    const int r1 = tokR[2 * t],     r2 = tokR[2 * t + 1];
    const size_t rowA = (size_t)(base[e1] + r1) * D_DIM;
    const size_t rowB = (size_t)(base[e2] + r2) * D_DIM;
    const size_t rowS = (size_t)(base[NE] + t)  * D_DIM;

    const int d = threadIdx.x * 4;
    float4 a = *(const float4*)(y + rowA + d);
    float4 b = *(const float4*)(y + rowB + d);
    float4 s = *(const float4*)(y + rowS + d);
    float4 o;
    o.x = a.x + b.x + s.x;
    o.y = a.y + b.y + s.y;
    o.z = a.z + b.z + s.z;
    o.w = a.w + b.w + s.w;
    *(float4*)(out + (size_t)t * D_DIM + d) = o;
}

// ============================================================
// host launcher
// ============================================================
extern "C" void kernel_launch(void* const* d_in, const int* in_sizes, int n_in,
                              void* d_out, int out_size, void* d_ws, size_t ws_size,
                              hipStream_t stream)
{
    const float* x   = (const float*)d_in[0];
    const float* Wg  = (const float*)d_in[1];
    const float* W1  = (const float*)d_in[2];
    const float* b1  = (const float*)d_in[3];
    const float* W2  = (const float*)d_in[4];
    const float* b2  = (const float*)d_in[5];
    const float* Ws1 = (const float*)d_in[6];
    const float* bs1 = (const float*)d_in[7];
    const float* Ws2 = (const float*)d_in[8];
    const float* bs2 = (const float*)d_in[9];
    float* out = (float*)d_out;

    char* ws = (char*)d_ws;
    int*    cnt     = (int*)(ws + 0);
    int*    base    = (int*)(ws + 256);
    int*    tokList = (int*)(ws + 512);                               // 8*4096 int
    float*  vList   = (float*)(ws + 512 + 4 * NE * CAP);              // 8*4096 f32
    int*    tokE    = (int*)(ws + 512 + 8 * NE * CAP);                // 2*T int
    int*    tokR    = (int*)(ws + 512 + 8 * NE * CAP + 8 * NTOK);     // 2*T int
    char*   p       = ws + 512 + 8 * NE * CAP + 16 * NTOK;            // 328192 (256-aligned)
    __bf16* xb      = (__bf16*)p;                                     // T x D bf16
    p += (size_t)NTOK * D_DIM * 2;
    __bf16* h       = (__bf16*)p;                                     // 12288 x 4096 bf16
    p += (size_t)TOTROWS * FF_DIM * 2;
    float*  y       = (float*)p;                                      // 12288 x 1024 f32

    hipLaunchKernelGGL(zero_kernel, dim3(1), dim3(32), 0, stream, cnt);
    hipLaunchKernelGGL(xcvt_kernel, dim3((NTOK * D_DIM) / (256 * 8)), dim3(256), 0, stream, x, xb);
    hipLaunchKernelGGL(gate_kernel, dim3(NTOK / 8), dim3(256), 0, stream,
                       x, Wg, cnt, tokList, vList, tokE, tokR);
    hipLaunchKernelGGL(scan_kernel, dim3(1), dim3(1), 0, stream, cnt, base);
    hipLaunchKernelGGL(ffn1_kernel, dim3(FF_DIM / BN, CAP / BM, NE + 1), dim3(256), 0, stream,
                       xb, W1, b1, Ws1, bs1, cnt, base, tokList, h);
    hipLaunchKernelGGL(ffn2_kernel, dim3(D_DIM / BN, CAP / BM, NE + 1), dim3(256), 0, stream,
                       h, W2, b2, Ws2, bs2, cnt, base, vList, y);
    hipLaunchKernelGGL(combine_kernel, dim3(NTOK), dim3(256), 0, stream,
                       y, base, tokE, tokR, out);
}
